// AAConv2d_17179869365
// MI455X (gfx1250) — matrix operand order repaired and checked by reading, not compiled, not run on hardware
//
#include <hip/hip_runtime.h>

// ---------------------------------------------------------------------------
// AAConv2d for MI455X (gfx1250, wave32, WMMA bf16 16x16x32)
// B=8, C=128, H=W=32 (HW=1024), OUT=256, DK=DV=128, NH=8, dkh=dvh=16
// Strategy: pre-pack every WMMA operand into fragment-native bf16 layout so
// all GEMM inner loops are {global_load_b128 x2 / ds_load_b128 x2} + v_wmma.
// Conv B tiles are staged to LDS with async-to-LDS double buffering.
// ---------------------------------------------------------------------------

typedef __attribute__((ext_vector_type(16))) __bf16 v16bf;
typedef __attribute__((ext_vector_type(8)))  float  v8f;
typedef int v4i __attribute__((vector_size(16)));
typedef __attribute__((address_space(1))) v4i* gv4i_p;  // global v4i*
typedef __attribute__((address_space(3))) v4i* sv4i_p;  // LDS v4i*

#define HWSZ 1024
#define CIN  128

#if __has_builtin(__builtin_amdgcn_global_load_async_to_lds_b128) && \
    __has_builtin(__builtin_amdgcn_s_wait_asynccnt)
#define USE_ASYNC_LDS 1
#else
#define USE_ASYNC_LDS 0
#endif

__device__ __forceinline__ v8f wmma_bf16(v16bf a, v16bf b, v8f c) {
  return __builtin_amdgcn_wmma_f32_16x16x32_bf16(false, a, false, b,
                                                 (short)0, c, false, false);
}

// A-fragment K index for element slot p (0..15): contiguous k = 16*hi + p (matches B packs).
__device__ __forceinline__ int a_kidx(int hi, int p) {
  return (hi ? 16 : 0) + p;
}

// ---------------------------------------------------------------------------
// Pack kernels (run once per call; elementwise gathers + cvt live here, not in
// the hot GEMM loops).
// ---------------------------------------------------------------------------

// x (B,C,32,32) f32 -> zero-padded bf16 image (B,C,34,34)
__global__ void pad_x_kernel(const float* __restrict__ x, __bf16* __restrict__ xp) {
  const int bc = blockIdx.x;  // 0..1023
  for (int i = threadIdx.x; i < 34 * 34; i += blockDim.x) {
    const int y = i / 34, xx = i % 34;
    float v = 0.0f;
    if (y >= 1 && y <= 32 && xx >= 1 && xx <= 32)
      v = x[(size_t)bc * HWSZ + (y - 1) * 32 + (xx - 1)];
    xp[(size_t)bc * 1156 + i] = (__bf16)v;
  }
}

// Generic A-operand pack: A[m][k] = src[(m*ldm + k)*kstep + pos]
// grid (kchunks, mtiles, npos), block 32
__global__ void pack_a_kernel(const float* __restrict__ src, __bf16* __restrict__ dst,
                              int mtiles, int kchunks, int ldm, int kstep) {
  const int chunk = blockIdx.x, mt = blockIdx.y, pos = blockIdx.z;
  const int lane = threadIdx.x & 31, hi = lane >> 4, ln = lane & 15;
  v16bf f;
#pragma unroll
  for (int p = 0; p < 16; ++p) {
    const int ka = chunk * 32 + a_kidx(hi, p);
    f[p] = (__bf16)src[((size_t)(mt * 16 + ln) * ldm + ka) * kstep + pos];
  }
  *(v16bf*)&dst[((((size_t)pos * kchunks + chunk) * mtiles + mt)) * 512 + lane * 16] = f;
}

// qkv B pack: B[k][n] = x[b][c0+k][nt*16+n]; grid (4, 64, 8)
__global__ void pack_bx_kernel(const float* __restrict__ x, __bf16* __restrict__ dst) {
  const int chunk = blockIdx.x, nt = blockIdx.y, b = blockIdx.z;
  const int lane = threadIdx.x & 31, hi = lane >> 4, ln = lane & 15;
  v16bf f;
#pragma unroll
  for (int p = 0; p < 16; ++p) {
    const int kb = chunk * 32 + (hi ? 16 : 0) + p;
    f[p] = (__bf16)x[((size_t)b * CIN + kb) * HWSZ + nt * 16 + ln];
  }
  *(v16bf*)&dst[(((size_t)b * 4 + chunk) * 64 + nt) * 512 + lane * 16] = f;
}

// conv B pack from padded image: grid (36 = (kh*3+kw)*4+chunk, 64 nt, 8 b)
__global__ void pack_convB_kernel(const __bf16* __restrict__ xp, __bf16* __restrict__ dst) {
  const int it = blockIdx.x, nt = blockIdx.y, b = blockIdx.z;
  const int pos = it >> 2, chunk = it & 3;
  const int kh = pos / 3, kw = pos % 3;
  const int lane = threadIdx.x & 31, hi = lane >> 4, ln = lane & 15;
  const int pix = nt * 16 + ln, py = pix >> 5, px = pix & 31;
  v16bf f;
#pragma unroll
  for (int p = 0; p < 16; ++p) {
    const int c = chunk * 32 + (hi ? 16 : 0) + p;
    f[p] = xp[((size_t)(b * CIN + c) * 34 + py + kh) * 34 + (px + kw)];
  }
  *(v16bf*)&dst[(((size_t)(b * 64 + nt)) * 36 + it) * 512 + lane * 16] = f;
}

// attention operand packs from qkv buffer: grid (64 tile, 64 bn, 3 kind)
// kind 0: Aq[m=p][k=d]   kind 1: Bk[k=d][n=keypos]   kind 2: Bv[k=keypos][n=d]
__global__ void pack_attn_kernel(const float* __restrict__ qkv,
                                 __bf16* __restrict__ aq,
                                 __bf16* __restrict__ bk,
                                 __bf16* __restrict__ bv) {
  const int tile = blockIdx.x, bn = blockIdx.y, kind = blockIdx.z;
  const int b = bn >> 3, nh = bn & 7;
  const int lane = threadIdx.x & 31, hi = lane >> 4, ln = lane & 15;
  v16bf f;
  if (kind == 0) {
    const float* qb = qkv + ((size_t)b * 384 + nh * 16) * HWSZ;
#pragma unroll
    for (int p = 0; p < 16; ++p) {
      const int ka = a_kidx(hi, p);
      f[p] = (ka < 16) ? (__bf16)qb[(size_t)ka * HWSZ + tile * 16 + ln] : (__bf16)0.0f;
    }
    *(v16bf*)&aq[(((size_t)bn * 64 + tile) * 32 + lane) * 16] = f;
  } else if (kind == 1) {
    const float* kb = qkv + ((size_t)b * 384 + 128 + nh * 16) * HWSZ;
#pragma unroll
    for (int p = 0; p < 16; ++p) {
      const int kk = (hi ? 16 : 0) + p;
      f[p] = (kk < 16) ? (__bf16)kb[(size_t)kk * HWSZ + tile * 16 + ln] : (__bf16)0.0f;
    }
    *(v16bf*)&bk[(((size_t)bn * 64 + tile) * 32 + lane) * 16] = f;
  } else {
    const float* vb = qkv + ((size_t)b * 384 + 256 + nh * 16) * HWSZ;
#pragma unroll
    for (int p = 0; p < 16; ++p) {
      const int kk = (hi ? 16 : 0) + p;
      f[p] = (kk < 16) ? (__bf16)vb[(size_t)ln * HWSZ + tile * 16 + kk] : (__bf16)0.0f;
    }
    *(v16bf*)&bv[(((size_t)bn * 64 + tile) * 32 + lane) * 16] = f;
  }
}

// ---------------------------------------------------------------------------
// qkv 1x1 GEMM (packed operands): grid (64 nt, 24 mt, 8 b), block 32
// ---------------------------------------------------------------------------
__global__ void qkv_gemm_kernel(const __bf16* __restrict__ Ap,
                                const __bf16* __restrict__ Bp,
                                const float* __restrict__ bias,
                                float* __restrict__ qkv) {
  const int nt = blockIdx.x, mt = blockIdx.y, b = blockIdx.z;
  const int lane = threadIdx.x & 31, hi = lane >> 4, ln = lane & 15;
  v8f acc = {};
#pragma unroll
  for (int chunk = 0; chunk < 4; ++chunk) {
    v16bf A = *(const v16bf*)&Ap[((size_t)chunk * 24 + mt) * 512 + lane * 16];
    v16bf Bf = *(const v16bf*)&Bp[(((size_t)b * 4 + chunk) * 64 + nt) * 512 + lane * 16];
    acc = wmma_bf16(A, Bf, acc);
  }
#pragma unroll
  for (int r = 0; r < 8; ++r) {
    const int o = mt * 16 + r + 8 * hi;
    float v = acc[r] + bias[o];
    if (o < 128) v *= 0.25f;  // dkh^-0.5 folded into q (content + rel logits)
    qkv[((size_t)b * 384 + o) * HWSZ + nt * 16 + ln] = v;
  }
}

// ---------------------------------------------------------------------------
// 3x3 conv implicit GEMM: grid (64 nt, 8 b), block 256 (8 waves = 8 ch-tiles).
// B tile (1KB, shared by all waves) staged to LDS with async double buffering.
// ---------------------------------------------------------------------------
__global__ void conv_gemm_kernel(const __bf16* __restrict__ Apack,
                                 const __bf16* __restrict__ Bpack,
                                 const float* __restrict__ cb,
                                 float* __restrict__ out) {
  __shared__ __attribute__((aligned(64))) __bf16 Bs[2][512];
  const int nt = blockIdx.x, b = blockIdx.y;
  const int tid = threadIdx.x, wave = tid >> 5, lane = tid & 31;
  const int hi = lane >> 4, ln = lane & 15;
  const __bf16* Bbase = Bpack + ((size_t)(b * 64 + nt)) * 36 * 512;

  auto stage = [&](int it, int buf) {
    if (wave == 0) {
      const __bf16* g = Bbase + (size_t)it * 512 + lane * 8;  // 16B per lane
      __bf16* s = &Bs[buf][lane * 8];
#if USE_ASYNC_LDS
      __builtin_amdgcn_global_load_async_to_lds_b128((gv4i_p)g, (sv4i_p)s, 0, 0);
      __builtin_amdgcn_global_load_async_to_lds_b128((gv4i_p)g, (sv4i_p)s, 512, 0);
#else
      const uint4* g4 = (const uint4*)g;
      uint4* s4 = (uint4*)s;
      s4[0] = g4[0];
      s4[32] = g4[32];  // +512 bytes
#endif
    }
  };

  stage(0, 0);
  v8f acc = {};
  for (int it = 0; it < 36; ++it) {
    const int buf = it & 1;
    if (it + 1 < 36) {
      stage(it + 1, buf ^ 1);
#if USE_ASYNC_LDS
      if (wave == 0) __builtin_amdgcn_s_wait_asynccnt(2);  // current buf done
#endif
    } else {
#if USE_ASYNC_LDS
      if (wave == 0) __builtin_amdgcn_s_wait_asynccnt(0);
#endif
    }
    __syncthreads();
    v16bf A = *(const v16bf*)&Apack[((size_t)it * 8 + wave) * 512 + lane * 16];
    v16bf Bf = *(const v16bf*)&Bs[buf][lane * 16];
    acc = wmma_bf16(A, Bf, acc);
    __syncthreads();  // all waves done reading before buf is re-staged
  }
#pragma unroll
  for (int r = 0; r < 8; ++r) {
    const int o = wave * 16 + r + 8 * hi;
    out[((size_t)b * 256 + o) * HWSZ + nt * 16 + ln] = acc[r] + cb[o];
  }
}

// ---------------------------------------------------------------------------
// Relative-logit tables: qr_w/qr_h[bn, p, m] = sum_d q[bn,d,p]*rel[m,d]
// grid (4 m-tiles, 64 p-tiles, 64 b*nh), block 32
// ---------------------------------------------------------------------------
__global__ void qrel_kernel(const __bf16* __restrict__ aqP,
                            const float* __restrict__ relw,
                            const float* __restrict__ relh,
                            float* __restrict__ qrw,
                            float* __restrict__ qrh) {
  const int nt = blockIdx.x, pt = blockIdx.y, bn = blockIdx.z;
  const int lane = threadIdx.x & 31, hi = lane >> 4, ln = lane & 15;

  v16bf A = *(const v16bf*)&aqP[(((size_t)bn * 64 + pt) * 32 + lane) * 16];
  const int mcol = nt * 16 + ln;
  v16bf Bw, Bh;
#pragma unroll
  for (int p = 0; p < 16; ++p) {
    const int kb = (hi ? 16 : 0) + p;
    const bool ok = (kb < 16) && (mcol < 63);
    Bw[p] = ok ? (__bf16)relw[mcol * 16 + kb] : (__bf16)0.0f;
    Bh[p] = ok ? (__bf16)relh[mcol * 16 + kb] : (__bf16)0.0f;
  }
  v8f z = {};
  v8f aw = wmma_bf16(A, Bw, z);
  v8f ah = wmma_bf16(A, Bh, z);
#pragma unroll
  for (int r = 0; r < 8; ++r) {
    const int p = pt * 16 + r + 8 * hi;
    qrw[((size_t)bn * HWSZ + p) * 64 + mcol] = aw[r];
    qrh[((size_t)bn * HWSZ + p) * 64 + mcol] = ah[r];
  }
}

// ---------------------------------------------------------------------------
// Fused flash attention; writes the projection's B operand in fragment layout.
// grid (64 p-tiles, 64 b*nh), block 32
// ---------------------------------------------------------------------------
__global__ void attn_kernel(const __bf16* __restrict__ aqP,
                            const __bf16* __restrict__ bkP,
                            const __bf16* __restrict__ bvP,
                            const float* __restrict__ qrw,
                            const float* __restrict__ qrh,
                            __bf16* __restrict__ projB) {
  __shared__ __attribute__((aligned(64))) __bf16 Pt[16 * 16];
  const int pt = blockIdx.x, bn = blockIdx.y;
  const int b = bn >> 3, nh = bn & 7;
  const int lane = threadIdx.x & 31, hi = lane >> 4, ln = lane & 15;

  v16bf Aq = *(const v16bf*)&aqP[(((size_t)bn * 64 + pt) * 32 + lane) * 16];

  float mrun[8], lrun[8];
#pragma unroll
  for (int r = 0; r < 8; ++r) { mrun[r] = -1e30f; lrun[r] = 0.0f; }
  v8f o = {};

  for (int qt = 0; qt < 64; ++qt) {
    v16bf Bk = *(const v16bf*)&bkP[(((size_t)bn * 64 + qt) * 32 + lane) * 16];
    v8f z = {};
    v8f s = wmma_bf16(Aq, Bk, z);

    // relative logits via lookup tables
    const int qpos = qt * 16 + ln, y2 = qpos >> 5, x2 = qpos & 31;
    float sv[8];
#pragma unroll
    for (int r = 0; r < 8; ++r) {
      const int p = pt * 16 + r + 8 * hi;
      const int y1 = p >> 5, x1 = p & 31;
      sv[r] = s[r] + qrw[((size_t)bn * HWSZ + p) * 64 + (x2 - x1 + 31)]
                   + qrh[((size_t)bn * HWSZ + p) * 64 + (y2 - y1 + 31)];
    }

    // online softmax; row stats live per-lane thanks to C/D half-lane layout
    float pe[8];
#pragma unroll
    for (int r = 0; r < 8; ++r) {
      float mx = sv[r];
      mx = fmaxf(mx, __shfl_xor(mx, 1, 32));
      mx = fmaxf(mx, __shfl_xor(mx, 2, 32));
      mx = fmaxf(mx, __shfl_xor(mx, 4, 32));
      mx = fmaxf(mx, __shfl_xor(mx, 8, 32));
      const float mnew = fmaxf(mrun[r], mx);
      const float scale = __expf(mrun[r] - mnew);
      o[r] *= scale;
      pe[r] = __expf(sv[r] - mnew);
      float rs = pe[r];
      rs += __shfl_xor(rs, 1, 32);
      rs += __shfl_xor(rs, 2, 32);
      rs += __shfl_xor(rs, 4, 32);
      rs += __shfl_xor(rs, 8, 32);
      lrun[r] = lrun[r] * scale + rs;
      mrun[r] = mnew;
    }

    // P (C-layout) -> LDS -> A-layout for the PV WMMA
#pragma unroll
    for (int r = 0; r < 8; ++r)
      Pt[(r + 8 * hi) * 16 + ln] = (__bf16)pe[r];
    __syncthreads();
    v16bf Apv;
#pragma unroll
    for (int p = 0; p < 16; ++p) {
      const int ka = a_kidx(hi, p);
      Apv[p] = (ka < 16) ? Pt[ln * 16 + ka] : (__bf16)0.0f;
    }
    v16bf Bv = *(const v16bf*)&bvP[(((size_t)bn * 64 + qt) * 32 + lane) * 16];
    o = wmma_bf16(Apv, Bv, o);
    __syncthreads();
  }

  // finalize + faithful raw reshape (b,NH,HW,dvh)->(b,NH*16,HW), emitted
  // directly into the projection's B-fragment pack.
#pragma unroll
  for (int r = 0; r < 8; ++r) {
    const int m = r + 8 * hi;
    const int p = pt * 16 + m;
    const int i = p * 16 + ln;  // d = ln
    const int ch = nh * 16 + (i >> 10);
    const int pos = i & 1023;
    const int chunk = ch >> 5, kk = ch & 31;
    const int pnt = pos >> 4, n = pos & 15;
    const int plane = n + (kk & 16), pp = kk & 15;
    projB[((((size_t)b * 4 + chunk) * 64 + pnt) * 32 + plane) * 16 + pp] =
        (__bf16)(o[r] / lrun[r]);
  }
}

// ---------------------------------------------------------------------------
// Final projection: out[:,128:256] = attn_w @ attn + attn_b
// grid (64 nt, 8 mt, 8 b), block 32
// ---------------------------------------------------------------------------
__global__ void proj_gemm_kernel(const __bf16* __restrict__ Ap,
                                 const __bf16* __restrict__ Bp,
                                 const float* __restrict__ ab,
                                 float* __restrict__ out) {
  const int nt = blockIdx.x, mt = blockIdx.y, b = blockIdx.z;
  const int lane = threadIdx.x & 31, hi = lane >> 4, ln = lane & 15;
  v8f acc = {};
#pragma unroll
  for (int chunk = 0; chunk < 4; ++chunk) {
    v16bf A = *(const v16bf*)&Ap[((size_t)chunk * 8 + mt) * 512 + lane * 16];
    v16bf Bf = *(const v16bf*)&Bp[(((size_t)b * 4 + chunk) * 64 + nt) * 512 + lane * 16];
    acc = wmma_bf16(A, Bf, acc);
  }
#pragma unroll
  for (int r = 0; r < 8; ++r) {
    const int o = mt * 16 + r + 8 * hi;
    out[((size_t)b * 256 + 128 + o) * HWSZ + nt * 16 + ln] = acc[r] + ab[o];
  }
}

// ---------------------------------------------------------------------------
extern "C" void kernel_launch(void* const* d_in, const int* in_sizes, int n_in,
                              void* d_out, int out_size, void* d_ws, size_t ws_size,
                              hipStream_t stream) {
  const float* x      = (const float*)d_in[0];
  const float* conv_w = (const float*)d_in[1];
  const float* conv_b = (const float*)d_in[2];
  const float* qkv_w  = (const float*)d_in[3];
  const float* qkv_b  = (const float*)d_in[4];
  const float* attn_w = (const float*)d_in[5];
  const float* attn_b = (const float*)d_in[6];
  const float* relw   = (const float*)d_in[7];
  const float* relh   = (const float*)d_in[8];
  float* out = (float*)d_out;

  // workspace carve-up (~85 MB)
  char* w = (char*)d_ws;
  auto alloc = [&](size_t bytes) -> void* {
    void* p = (void*)w;
    w += (bytes + 255) & ~(size_t)255;
    return p;
  };
  __bf16* xpad  = (__bf16*)alloc((size_t)8 * 128 * 1156 * 2);
  __bf16* convA = (__bf16*)alloc((size_t)9 * 4 * 8 * 512 * 2);
  __bf16* qkvA  = (__bf16*)alloc((size_t)4 * 24 * 512 * 2);
  __bf16* projA = (__bf16*)alloc((size_t)4 * 8 * 512 * 2);
  __bf16* qkvB  = (__bf16*)alloc((size_t)8 * 4 * 64 * 512 * 2);
  __bf16* convB = (__bf16*)alloc((size_t)8 * 64 * 36 * 512 * 2);
  float*  qkvbf = (float*) alloc((size_t)8 * 384 * HWSZ * 4);
  __bf16* aqP   = (__bf16*)alloc((size_t)64 * 64 * 512 * 2);
  __bf16* bkP   = (__bf16*)alloc((size_t)64 * 64 * 512 * 2);
  __bf16* bvP   = (__bf16*)alloc((size_t)64 * 64 * 512 * 2);
  float*  qrw   = (float*) alloc((size_t)64 * HWSZ * 64 * 4);
  float*  qrh   = (float*) alloc((size_t)64 * HWSZ * 64 * 4);
  __bf16* projB = (__bf16*)alloc((size_t)8 * 4 * 64 * 512 * 2);

  // --- packing stage ---
  pad_x_kernel     <<<dim3(1024),        256, 0, stream>>>(x, xpad);
  pack_a_kernel    <<<dim3(4, 8, 9),     32,  0, stream>>>(conv_w, convA, 8, 4, 128, 9);
  pack_a_kernel    <<<dim3(4, 24, 1),    32,  0, stream>>>(qkv_w, qkvA, 24, 4, 128, 1);
  pack_a_kernel    <<<dim3(4, 8, 1),     32,  0, stream>>>(attn_w, projA, 8, 4, 128, 1);
  pack_bx_kernel   <<<dim3(4, 64, 8),    32,  0, stream>>>(x, qkvB);
  pack_convB_kernel<<<dim3(36, 64, 8),   32,  0, stream>>>(xpad, convB);

  // --- conv path ---
  conv_gemm_kernel <<<dim3(64, 8),       256, 0, stream>>>(convA, convB, conv_b, out);

  // --- attention path ---
  qkv_gemm_kernel  <<<dim3(64, 24, 8),   32,  0, stream>>>(qkvA, qkvB, qkv_b, qkvbf);
  pack_attn_kernel <<<dim3(64, 64, 3),   32,  0, stream>>>(qkvbf, aqP, bkP, bvP);
  qrel_kernel      <<<dim3(4, 64, 64),   32,  0, stream>>>(aqP, relw, relh, qrw, qrh);
  attn_kernel      <<<dim3(64, 64),      32,  0, stream>>>(aqP, bkP, bvP, qrw, qrh, projB);
  proj_gemm_kernel <<<dim3(64, 8, 8),    32,  0, stream>>>(projA, projB, attn_b, out);
}
